// Attention_76527727280452
// MI455X (gfx1250) — matrix operand order repaired and checked by reading, not compiled, not run on hardware
//
#include <hip/hip_runtime.h>
#include <hip/hip_bf16.h>

// ---------------------------------------------------------------------------
// Attention (b=8, n=1024, dim=768, heads=12, d_head=64), fp32 in/out,
// bf16 WMMA internals (v_wmma_f32_16x16x32_bf16, f32 accumulation).
//
// MI455X design: compute-bound (~65 GFLOP vs ~75MB unique data). All GEMM
// work on the WMMA pipe; operand reuse via the 192MB L2 (whole working set
// is L2-resident); 320KB LDS holds full 32x1024 score panels so softmax is
// one exact pass. GEMM waves use 32x64 tiles: 8 WMMAs per 12 b128 loads.
// ---------------------------------------------------------------------------

typedef __attribute__((ext_vector_type(16))) __bf16 v16bf;
typedef __attribute__((ext_vector_type(8)))  __bf16 v8bf;
typedef __attribute__((ext_vector_type(8)))  float  v8f;
typedef unsigned short u16;

#define DIMX   768
#define HEADS  12
#define DHEAD  64
#define BATCH  8
#define SEQ    1024
#define MROWS  (BATCH * SEQ)        // 8192
#define NQKV   (3 * DIMX)           // 2304

#define WMMA_BF16(a, b, c) \
  __builtin_amdgcn_wmma_f32_16x16x32_bf16(false, (a), false, (b), (short)0, (c), false, false)

// A-matrix fragment (16x32 bf16, MxK), contiguous K (matches B):
//   lanes 0-15 : row M=lane,    K = kb+0..15
//   lanes16-31 : row M=lane-16, K = kb+16..31
static __device__ inline v16bf load_a_frag(const __bf16* base, int ld, int row0,
                                           int kb, int lane) {
  const int half = lane >> 4, l = lane & 15;
  return *(const v16bf*)(base + (size_t)(row0 + l) * ld + kb + half * 16);
}

// B-matrix fragment (32x16 bf16, KxN), column-per-lane:
//   lanes 0-15 : col N=lane,    K = kb+0..15  (contiguous in memory along K)
//   lanes16-31 : col N=lane-16, K = kb+16..31
// base is laid out so that column n is a contiguous row of length ld.
static __device__ inline v16bf load_b_frag(const __bf16* base, int ld, int col0,
                                           int kb, int lane) {
  const int half = lane >> 4, n = lane & 15;
  return *(const v16bf*)(base + (size_t)(col0 + n) * ld + kb + half * 16);
}

// ---------------------------------------------------------------------------
__global__ void cvt_f32_bf16(const float* __restrict__ in, u16* __restrict__ outp,
                             int n) {
  __bf16* out = (__bf16*)outp;
  int i = blockIdx.x * blockDim.x + threadIdx.x;
  int stride = gridDim.x * blockDim.x;
  for (; i < n; i += stride) out[i] = (__bf16)in[i];
}

// ---------------------------------------------------------------------------
// QKV GEMM: [8192 x 768] x [768 x 2304] -> scatter to Q,K ([b,h,n,d]) and
// V transposed ([b,h,d,n]) so attention fragments are contiguous loads.
// Block = 256 threads (8 waves, 4x2 grid of 32x64 wave tiles) -> 128x128.
__global__ __launch_bounds__(256) void qkv_gemm(const u16* __restrict__ Xu,
                                                const u16* __restrict__ Wu,
                                                u16* __restrict__ Qu,
                                                u16* __restrict__ Ku,
                                                u16* __restrict__ VTu) {
  const __bf16* X = (const __bf16*)Xu;
  const __bf16* W = (const __bf16*)Wu;
  __bf16* Q  = (__bf16*)Qu;
  __bf16* Km = (__bf16*)Ku;
  __bf16* VT = (__bf16*)VTu;

  const int lane = threadIdx.x & 31;
  const int w    = threadIdx.x >> 5;              // 0..7
  const int row0 = blockIdx.x * 128 + (w >> 1) * 32;
  const int col0 = blockIdx.y * 128 + (w & 1) * 64;

  v8f acc[2][4] = {};
  for (int kb = 0; kb < DIMX; kb += 32) {
    if (kb + 64 <= DIMX - 32) {  // prefetch next-next K-step rows into cache
      __builtin_prefetch(X + (size_t)(row0 + lane) * DIMX + kb + 64, 0, 1);
      __builtin_prefetch(W + (size_t)(col0 + lane) * DIMX + kb + 64, 0, 1);
      __builtin_prefetch(W + (size_t)(col0 + 32 + lane) * DIMX + kb + 64, 0, 1);
    }
    v16bf a0 = load_a_frag(X, DIMX, row0,      kb, lane);
    v16bf a1 = load_a_frag(X, DIMX, row0 + 16, kb, lane);
#pragma unroll
    for (int jc = 0; jc < 4; ++jc) {
      v16bf bf = load_b_frag(W, DIMX, col0 + jc * 16, kb, lane);
      acc[0][jc] = WMMA_BF16(a0, bf, acc[0][jc]);
      acc[1][jc] = WMMA_BF16(a1, bf, acc[1][jc]);
    }
  }

  const int half = lane >> 4, n = lane & 15;
#pragma unroll
  for (int ti = 0; ti < 2; ++ti)
#pragma unroll
    for (int tj = 0; tj < 4; ++tj) {
      const int e   = col0 + tj * 16 + n;       // global output column [0,2304)
      const int sec = e / DIMX;                 // 0=q, 1=k, 2=v
      const int rem = e - sec * DIMX;
      const int h = rem >> 6, d = rem & 63;
#pragma unroll
      for (int j = 0; j < 8; ++j) {
        const int m  = row0 + ti * 16 + j + 8 * half;   // global row [0,8192)
        const int b  = m >> 10, nn = m & 1023;
        const __bf16 val = (__bf16)acc[ti][tj][j];
        const size_t bh = (size_t)(b * HEADS + h);
        if (sec == 0)      Q [(bh * SEQ + nn) * DHEAD + d] = val;
        else if (sec == 1) Km[(bh * SEQ + nn) * DHEAD + d] = val;
        else               VT[(bh * DHEAD + d) * SEQ + nn] = val;  // transposed
      }
    }
}

// ---------------------------------------------------------------------------
// Attention core. Grid: (96 bh, 32 q-tiles). Block: 64 threads (2 waves),
// each wave owns 16 query rows. Full 32x1024 score panel lives in LDS
// (128KB f32 scores + 64KB bf16 probs + 512B reductions <= 320KB CDNA5 LDS),
// so softmax is a single exact pass, no online rescaling.
__global__ __launch_bounds__(64) void attn_core(const u16* __restrict__ Qu,
                                                const u16* __restrict__ Ku,
                                                const u16* __restrict__ VTu,
                                                u16* __restrict__ Ou) {
  extern __shared__ char smem[];
  float*  S    = (float*)smem;                               // 32*1024 f32
  __bf16* P    = (__bf16*)(smem + 32 * SEQ * sizeof(float)); // 32*1024 bf16
  float*  redm = (float*)(smem + 32 * SEQ * 6);              // 64 f32
  float*  reds = redm + 64;                                  // 64 f32

  const __bf16* Q  = (const __bf16*)Qu;
  const __bf16* Km = (const __bf16*)Ku;
  const __bf16* VT = (const __bf16*)VTu;
  __bf16* O = (__bf16*)Ou;

  const int bh = blockIdx.x;
  const int qt = blockIdx.y;
  const int b  = bh / HEADS, h = bh % HEADS;
  const int lane = threadIdx.x & 31;
  const int wv   = threadIdx.x >> 5;
  const int half = lane >> 4, n = lane & 15;

  const __bf16* Qb = Q  + (size_t)bh * SEQ * DHEAD;   // [n, d]
  const __bf16* Kb = Km + (size_t)bh * SEQ * DHEAD;   // [n, d]
  const __bf16* Vb = VT + (size_t)bh * DHEAD * SEQ;   // [d, n] (transposed)

  const int q0 = qt * 32 + wv * 16;   // first query row for this wave

  // ---- Phase 1: S = Q K^T (unscaled, faithful to reference) ----
  v16bf qa0 = load_a_frag(Qb, DHEAD, q0, 0,  lane);
  v16bf qa1 = load_a_frag(Qb, DHEAD, q0, 32, lane);
  for (int nt = 0; nt < SEQ / 16; ++nt) {
    v16bf kb0 = load_b_frag(Kb, DHEAD, nt * 16, 0,  lane);
    v16bf kb1 = load_b_frag(Kb, DHEAD, nt * 16, 32, lane);
    v8f acc = {};
    acc = WMMA_BF16(qa0, kb0, acc);
    acc = WMMA_BF16(qa1, kb1, acc);
#pragma unroll
    for (int j = 0; j < 8; ++j)
      S[(wv * 16 + j + 8 * half) * SEQ + nt * 16 + n] = acc[j];
  }
  __syncthreads();

  // ---- Phase 2: exact softmax over full 1024-wide rows in LDS ----
  {
    const int t  = threadIdx.x;
    const int r  = t >> 1;              // local row 0..31
    const int c0 = (t & 1) * 512;       // half-row
    float* row = S + r * SEQ + c0;
    float mx = -3.4e38f;
    for (int c = 0; c < 512; ++c) mx = fmaxf(mx, row[c]);
    redm[t] = mx;
    __syncthreads();
    mx = fmaxf(redm[r * 2], redm[r * 2 + 1]);
    float sum = 0.f;
    for (int c = 0; c < 512; ++c) { float e = __expf(row[c] - mx); row[c] = e; sum += e; }
    reds[t] = sum;
    __syncthreads();
    const float inv = 1.0f / (reds[r * 2] + reds[r * 2 + 1]);
    __bf16* prow = P + r * SEQ + c0;
    for (int c = 0; c < 512; ++c) prow[c] = (__bf16)(row[c] * inv);
  }
  __syncthreads();

  // ---- Phase 3: O = P V  (A from LDS, B = contiguous rows of V^T) ----
  for (int dt = 0; dt < DHEAD / 16; ++dt) {
    v8f acc = {};
    for (int kb = 0; kb < SEQ; kb += 32) {
      v16bf a  = load_a_frag((const __bf16*)P, SEQ, wv * 16, kb, lane);
      v16bf bb = load_b_frag(Vb, SEQ, dt * 16, kb, lane);
      acc = WMMA_BF16(a, bb, acc);
    }
#pragma unroll
    for (int j = 0; j < 8; ++j) {
      const int m = q0 + j + 8 * half;  // query row within sequence
      O[((size_t)(b * SEQ + m)) * DIMX + h * DHEAD + dt * 16 + n] = (__bf16)acc[j];
    }
  }
}

// ---------------------------------------------------------------------------
// Output projection: [8192 x 768] x [768 x 768] + bias -> f32 result.
// Same 8-wave / 32x64-per-wave tiling as the QKV GEMM.
__global__ __launch_bounds__(256) void proj_gemm(const u16* __restrict__ Ou,
                                                 const u16* __restrict__ Wu,
                                                 const float* __restrict__ bias,
                                                 float* __restrict__ out) {
  const __bf16* O = (const __bf16*)Ou;
  const __bf16* W = (const __bf16*)Wu;

  const int lane = threadIdx.x & 31;
  const int w    = threadIdx.x >> 5;
  const int row0 = blockIdx.x * 128 + (w >> 1) * 32;
  const int col0 = blockIdx.y * 128 + (w & 1) * 64;

  v8f acc[2][4] = {};
  for (int kb = 0; kb < DIMX; kb += 32) {
    if (kb + 64 <= DIMX - 32) {
      __builtin_prefetch(O + (size_t)(row0 + lane) * DIMX + kb + 64, 0, 1);
      __builtin_prefetch(W + (size_t)(col0 + lane) * DIMX + kb + 64, 0, 1);
      __builtin_prefetch(W + (size_t)(col0 + 32 + lane) * DIMX + kb + 64, 0, 1);
    }
    v16bf a0 = load_a_frag(O, DIMX, row0,      kb, lane);
    v16bf a1 = load_a_frag(O, DIMX, row0 + 16, kb, lane);
#pragma unroll
    for (int jc = 0; jc < 4; ++jc) {
      v16bf bf = load_b_frag(W, DIMX, col0 + jc * 16, kb, lane);
      acc[0][jc] = WMMA_BF16(a0, bf, acc[0][jc]);
      acc[1][jc] = WMMA_BF16(a1, bf, acc[1][jc]);
    }
  }

  const int half = lane >> 4, n = lane & 15;
#pragma unroll
  for (int ti = 0; ti < 2; ++ti)
#pragma unroll
    for (int tj = 0; tj < 4; ++tj) {
      const int e = col0 + tj * 16 + n;
      const float bv = bias[e];
#pragma unroll
      for (int j = 0; j < 8; ++j) {
        const int m = row0 + ti * 16 + j + 8 * half;
        out[(size_t)m * DIMX + e] = acc[ti][tj][j] + bv;
      }
    }
}

// ---------------------------------------------------------------------------
extern "C" void kernel_launch(void* const* d_in, const int* in_sizes, int n_in,
                              void* d_out, int out_size, void* d_ws, size_t ws_size,
                              hipStream_t stream) {
  (void)in_sizes; (void)n_in; (void)out_size; (void)ws_size;
  const float* x     = (const float*)d_in[0];
  const float* w_qkv = (const float*)d_in[1];
  const float* w_fc  = (const float*)d_in[2];
  const float* b_fc  = (const float*)d_in[3];
  float* out = (float*)d_out;

  char* ws = (char*)d_ws;
  size_t off = 0;
  u16* xb    = (u16*)(ws + off); off += (size_t)MROWS * DIMX * 2;  // x in bf16
  u16* wqkvb = (u16*)(ws + off); off += (size_t)NQKV  * DIMX * 2;  // w_qkv bf16
  u16* wfcb  = (u16*)(ws + off); off += (size_t)DIMX  * DIMX * 2;  // w_fc bf16
  u16* qb    = (u16*)(ws + off); off += (size_t)MROWS * DIMX * 2;  // Q [b,h,n,d]
  u16* kbuf  = (u16*)(ws + off); off += (size_t)MROWS * DIMX * 2;  // K [b,h,n,d]
  u16* vtb   = (u16*)(ws + off); off += (size_t)MROWS * DIMX * 2;  // V^T [b,h,d,n]
  u16* ob    = (u16*)(ws + off); off += (size_t)MROWS * DIMX * 2;  // attn out bf16

  // 1) fp32 -> bf16 conversions (inputs + weights)
  cvt_f32_bf16<<<512, 256, 0, stream>>>(x,     xb,    MROWS * DIMX);
  cvt_f32_bf16<<<128, 256, 0, stream>>>(w_qkv, wqkvb, NQKV  * DIMX);
  cvt_f32_bf16<<<64,  256, 0, stream>>>(w_fc,  wfcb,  DIMX  * DIMX);

  // 2) QKV projection GEMM (scatter epilogue to Q, K, V^T)
  qkv_gemm<<<dim3(MROWS / 128, NQKV / 128), 256, 0, stream>>>(xb, wqkvb, qb, kbuf, vtb);

  // 3) Attention: 96 (b,h) x 32 q-tiles; 192.5KB dynamic LDS per block
  const size_t attn_lds = 32 * SEQ * sizeof(float)      // scores f32
                        + 32 * SEQ * sizeof(u16)        // probs bf16
                        + 128 * sizeof(float);          // reductions
  attn_core<<<dim3(BATCH * HEADS, SEQ / 32), 64, attn_lds, stream>>>(qb, kbuf, vtb, ob);

  // 4) Output projection + bias -> f32
  proj_gemm<<<dim3(MROWS / 128, DIMX / 128), 256, 0, stream>>>(ob, wfcb, b_fc, out);
}
